// SharedLSTM_40716289966344
// MI455X (gfx1250) — compile-verified, matrix operand order checked by reading, not run on hardware
//
#include <hip/hip_runtime.h>
#include <hip/hip_bf16.h>

typedef __bf16        v16bf __attribute__((ext_vector_type(16)));
typedef float         v8f   __attribute__((ext_vector_type(8)));
typedef unsigned int  v8u   __attribute__((ext_vector_type(8)));
typedef unsigned int  u32x4 __attribute__((ext_vector_type(4)));
typedef int           i32x4 __attribute__((ext_vector_type(4)));
typedef int           i32x8 __attribute__((ext_vector_type(8)));

#define B_     64
#define S_     512
#define D_     512
#define FOURD  2048

// Tensor Data Mover availability (device pass only; host pass parses fallback)
#if defined(__HIP_DEVICE_COMPILE__) && __has_builtin(__builtin_amdgcn_tensor_load_to_lds)
#define HAVE_TDM 1
#else
#define HAVE_TDM 0
#endif

// workspace layout (bytes)
static constexpr size_t XPROJ_ELEMS = (size_t)S_ * B_ * FOURD;         // 67,108,864 fp32
static constexpr size_t XPROJ_BYTES = XPROJ_ELEMS * sizeof(float);     // 256 MB
static constexpr size_t WT_BYTES    = (size_t)FOURD * D_ * 2;          // 2 MB bf16 (transposed)
static constexpr size_t HBF_BYTES   = 2ull * B_ * D_ * 2;              // h double buffer, bf16
static constexpr size_t C_BYTES     = (size_t)B_ * D_ * 4;             // cell state fp32

__device__ __forceinline__ float sigmoidf_(float x) {
  return 1.0f / (1.0f + __expf(-x));
}

// 16x32 bf16 WMMA operand fragment (A or B, symmetric layout), 4x 8-byte loads.
// p = &tile[row][k0] (row = lane&15), half = lane>>4.
__device__ __forceinline__ v16bf frag_ld(const __bf16* p, int half) {
  v8u r = {};
  const int ho = half * 8;
#pragma unroll
  for (int g = 0; g < 4; ++g) {
    const int kb = ((g >> 1) << 4) + ((g & 1) << 2) + ho;
    const uint2 d = *(const uint2*)(p + kb);
    r[2 * g + 0] = d.x;
    r[2 * g + 1] = d.y;
  }
  return __builtin_bit_cast(v16bf, r);
}

#if HAVE_TDM
// Issue a 2D bf16 tile DMA (global row-major -> LDS, optional row padding).
// D# group0: count=1 | lds_addr | 57b global_addr | type=2.
// D# group1: data_size=2B, pad_enable, pad_interval/amount, tensor/tile dims, dim0 stride.
// Dims/strides in elements (data_size units). Issued per-wave; EXEC ignored.
__device__ __forceinline__ void tdm_load_2d_bf16(
    unsigned lds_off, const void* gptr,
    unsigned tile_d0, unsigned tile_d1,
    unsigned tensor_d0, unsigned tensor_d1, unsigned stride_d0,
    unsigned pad_interval_code, unsigned pad_amount_code)
{
  const unsigned long long ga = (unsigned long long)(uintptr_t)gptr;
  u32x4 g0;
  g0.x = 1u;                                    // count=1 (valid), user mode
  g0.y = lds_off;                               // LDS byte address
  g0.z = (unsigned)ga;                          // global_addr[31:0]
  g0.w = (unsigned)((ga >> 32) & 0x01FFFFFFu)   // global_addr[56:32]
       | 0x80000000u;                           // type = 2 ("image")
  i32x8 g1;
  g1[0] = (int)((1u << 16)                      // data_size: 2 bytes
              | (1u << 20)                      // pad_enable
              | (pad_interval_code << 22)       // 2^(n+1) DWORDs between pads
              | (pad_amount_code << 25));       // (n+1) DWORDs of pad
  g1[1] = (int)((tensor_d0 & 0xFFFFu) << 16);                               // dim0[15:0]
  g1[2] = (int)(((tensor_d0 >> 16) & 0xFFFFu) | ((tensor_d1 & 0xFFFFu) << 16));
  g1[3] = (int)(((tensor_d1 >> 16) & 0xFFFFu) | ((tile_d0 & 0xFFFFu) << 16));
  g1[4] = (int)(tile_d1 & 0xFFFFu);             // tile_dim1 (tile_dim2 = 0)
  g1[5] = (int)stride_d0;                       // tensor_dim0_stride[31:0]
  g1[6] = 0;
  g1[7] = 0;
  const i32x4 z4 = {0, 0, 0, 0};
#if __clang_major__ >= 23
  const i32x8 z8 = {0, 0, 0, 0, 0, 0, 0, 0};
  __builtin_amdgcn_tensor_load_to_lds(g0, g1, z4, z4, z8, 0);
#else
  __builtin_amdgcn_tensor_load_to_lds(g0, g1, z4, z4, 0);
#endif
}
#endif  // HAVE_TDM

// ---------------------------------------------------------------------------
// Kernel 0: prep. Transpose W,U (fp32 [D][4D]) -> bf16 [4D][D]; zero h/c state
// and the grid-barrier counter (re-zeroed every launch: graph replays reuse
// d_ws without re-poisoning).
// ---------------------------------------------------------------------------
__global__ __launch_bounds__(256) void lstm_prep(
    const float* __restrict__ W, const float* __restrict__ U,
    __bf16* __restrict__ Wt, __bf16* __restrict__ Ut,
    __bf16* __restrict__ hbf, float* __restrict__ cbuf,
    unsigned* __restrict__ cnt)
{
  const int n = blockIdx.x;            // 0..2047: transposed row
  const int tid = threadIdx.x;
  for (int k = tid; k < D_; k += 256) {
    Wt[(size_t)n * D_ + k] = (__bf16)W[(size_t)k * FOURD + n];
    Ut[(size_t)n * D_ + k] = (__bf16)U[(size_t)k * FOURD + n];
  }
  const int gidx = blockIdx.x * 256 + tid;
  if (gidx < 2 * B_ * D_) hbf[gidx] = (__bf16)0.0f;   // h0 = 0 (both buffers)
  if (gidx < B_ * D_)     cbuf[gidx] = 0.0f;          // c0 = 0
  if (gidx == 0)          *cnt = 0u;
}

// ---------------------------------------------------------------------------
// Kernel 1: x_proj[s][b][4D] = cas@W + soc@U + b.  LDS-tiled bf16 WMMA GEMM.
// Block tile 64(M) x 128(N), K-chunk 32. B-tiles staged by the Tensor Data
// Mover (pad_enable -> conflict-free 36-bf16 LDS row stride) overlapped with
// the VALU fp32->bf16 A-tile conversion. Wave w: M-tile (w&3), N-tiles
// (w>>2)*4..+3; all operand fragments loaded before the 8-WMMA burst.
// ---------------------------------------------------------------------------
__global__ __launch_bounds__(256) void xproj_gemm(
    const float* __restrict__ cas, const float* __restrict__ soc,
    const __bf16* __restrict__ Wt, const __bf16* __restrict__ Ut,
    const float* __restrict__ bias, float* __restrict__ xproj)
{
  __shared__ __bf16 sAc[64][36];
  __shared__ __bf16 sAs[64][36];
  __shared__ __bf16 sBw[128][36];    // 32 data + 4 pad bf16 per row (TDM padding)
  __shared__ __bf16 sBu[128][36];

  const int tid  = threadIdx.x;
  const int lane = tid & 31;
  const int wid  = tid >> 5;
  const int half = lane >> 4;
  const int nn   = lane & 15;
  const int mt   = wid & 3;
  const int nh   = wid >> 2;

  const int bm = blockIdx.y;   // 0..511 (M blocks of 64 input rows b*S+s)
  const int bn = blockIdx.x;   // 0..15  (N blocks of 128 gate cols)

  v8f acc[4] = {};

  const int arow  = tid >> 2;          // 0..63
  const int akofs = (tid & 3) * 8;     // 0,8,16,24
#if !HAVE_TDM
  const int brow  = tid >> 1;          // 0..127
  const int bkofs = (tid & 1) * 16;    // 0,16
#endif

  for (int kc = 0; kc < D_; kc += 32) {
    __syncthreads();
#if HAVE_TDM
    if (wid == 0) {    // one DMA per B tile: 128 rows x 32 bf16, row pad 16->18 DW
      tdm_load_2d_bf16((unsigned)(uintptr_t)&sBw[0][0],
                       Wt + (size_t)(bn * 128) * D_ + kc,
                       /*tile*/ 32, 128, /*tensor*/ 512, 2048, /*stride*/ 512,
                       /*interval 16DW*/ 3, /*pad 2DW*/ 1);
      tdm_load_2d_bf16((unsigned)(uintptr_t)&sBu[0][0],
                       Ut + (size_t)(bn * 128) * D_ + kc,
                       32, 128, 512, 2048, 512, 3, 1);
    }
#else
    {   // fallback: manual bf16 copy of B tiles
      const size_t gr = (size_t)(bn * 128 + brow) * D_ + kc + bkofs;
      const unsigned* gw = (const unsigned*)(Wt + gr);
      const unsigned* gu = (const unsigned*)(Ut + gr);
      unsigned* lw = (unsigned*)&sBw[brow][bkofs];
      unsigned* lu = (unsigned*)&sBu[brow][bkofs];
#pragma unroll
      for (int i = 0; i < 8; ++i) { lw[i] = gw[i]; lu[i] = gu[i]; }
    }
#endif
    {   // A tiles: fp32 global -> bf16 LDS (overlaps the TDM transfer)
      const size_t gr = (size_t)(bm * 64 + arow) * D_ + kc + akofs;
#pragma unroll
      for (int i = 0; i < 8; ++i) {
        sAc[arow][akofs + i] = (__bf16)cas[gr + i];
        sAs[arow][akofs + i] = (__bf16)soc[gr + i];
      }
    }
#if HAVE_TDM
    if (wid == 0) __builtin_amdgcn_s_wait_tensorcnt(0);
#endif
    __syncthreads();

    // load all operand fragments first, then issue the WMMA burst
    const v16bf a_c = frag_ld(&sAc[mt * 16 + nn][0], half);
    const v16bf a_s = frag_ld(&sAs[mt * 16 + nn][0], half);
    v16bf bw[4], bu[4];
#pragma unroll
    for (int j = 0; j < 4; ++j) {
      const int nrow = (nh * 4 + j) * 16 + nn;
      bw[j] = frag_ld(&sBw[nrow][0], half);
      bu[j] = frag_ld(&sBu[nrow][0], half);
    }
#pragma unroll
    for (int j = 0; j < 4; ++j)
      acc[j] = __builtin_amdgcn_wmma_f32_16x16x32_bf16(
          false, a_c, false, bw[j], (short)0, acc[j], false, false);
#pragma unroll
    for (int j = 0; j < 4; ++j)
      acc[j] = __builtin_amdgcn_wmma_f32_16x16x32_bf16(
          false, a_s, false, bu[j], (short)0, acc[j], false, false);
  }

  // Store with [b][s] -> [s][b] row remap, + bias.
  // C/D layout: VGPR r, lane l -> M = r + 8*(l>>4), N = l&15.
#pragma unroll
  for (int j = 0; j < 4; ++j) {
    const int col = bn * 128 + (nh * 4 + j) * 16 + nn;
    const float bv = bias[col];
#pragma unroll
    for (int r = 0; r < 8; ++r) {
      const int ib = bm * 64 + mt * 16 + r + 8 * half;   // input row = b*S + s
      const int bb = ib >> 9;
      const int ss = ib & (S_ - 1);
      xproj[((size_t)ss * B_ + bb) * FOURD + col] = acc[j][r] + bv;
    }
  }
}

// ---------------------------------------------------------------------------
// Kernel 2: persistent LSTM recurrence. 16 WGs; WG wg owns D-columns
// [wg*32,+32) of all four gates. V slice (128x512 bf16, LDS-resident all 512
// steps). Each step the 64 KB bf16 h matrix is DMA'd into LDS by the TDM
// (row pad 256->260 DW = conflict-free 520-bf16 stride). Wave w: M-tile
// (w&3), col sub-tile (w>>2); holds all 4 gate accumulators so the gate
// nonlinearity + cell update fuse in-register. Cross-WG sync: monotonic
// atomic barrier (+ s_cluster_barrier, NOP outside cluster dispatch).
// ---------------------------------------------------------------------------
__global__ __launch_bounds__(256) void lstm_persistent(
    const float* __restrict__ xproj, const float* __restrict__ V,
    __bf16* __restrict__ hbf, float* __restrict__ cbuf,
    float* __restrict__ out, unsigned* __restrict__ cnt)
{
  extern __shared__ __bf16 smem[];
  constexpr int SVP = 520;                  // padded row stride (bf16)
  __bf16* sV = smem;                        // [128][520] V slice
#if HAVE_TDM
  __bf16* sH = smem + 128 * SVP;            // [64][520]  h tile (TDM dest)
#endif

  const int wg   = blockIdx.x;              // 0..15
  const int tid  = threadIdx.x;
  const int lane = tid & 31;
  const int wid  = tid >> 5;
  const int half = lane >> 4;
  const int nn   = lane & 15;
  const int mt   = wid & 3;
  const int sub  = wid >> 2;

  {   // one-time: convert this WG's V columns fp32 -> bf16 into LDS [n][k]
    const int n    = tid >> 1;              // local gate-col 0..127
    const int ks   = (tid & 1) * 256;
    const int gate = n >> 5;
    const int gcol = gate * D_ + wg * 32 + (n & 31);
    for (int k = ks; k < ks + 256; ++k)
      sV[n * SVP + k] = (__bf16)V[(size_t)k * FOURD + gcol];
  }
  __syncthreads();

  const size_t HT_OFF = (size_t)B_ * S_ * D_;
  const size_t CT_OFF = HT_OFF + (size_t)B_ * D_;
  const int dcol = wg * 32 + sub * 16 + nn;

  for (int t = 0; t < S_; ++t) {
    const __bf16* hread  = hbf + (size_t)(t & 1) * B_ * D_;
    __bf16*       hwrite = hbf + (size_t)((t + 1) & 1) * B_ * D_;

#if HAVE_TDM
    if (wid == 0) {   // DMA h(t): 64 rows x 512 bf16, row pad 256->260 DW
      tdm_load_2d_bf16((unsigned)(uintptr_t)sH, hread,
                       /*tile*/ 512, 64, /*tensor*/ 512, 64, /*stride*/ 512,
                       /*interval 256DW*/ 7, /*pad 4DW*/ 3);
      __builtin_amdgcn_s_wait_tensorcnt(0);
    }
    __syncthreads();
    const __bf16* arow_p = sH + (size_t)(mt * 16 + nn) * SVP;
#else
    const __bf16* arow_p = hread + (size_t)(mt * 16 + nn) * D_;
    constexpr int ARS = D_;
#endif

    v8f acc[4] = {};
#pragma unroll 4
    for (int kc = 0; kc < D_; kc += 32) {
      const v16bf a = frag_ld(arow_p + kc, half);
      v16bf bfr[4];
#pragma unroll
      for (int j = 0; j < 4; ++j)
        bfr[j] = frag_ld(&sV[((j * 2 + sub) * 16 + nn) * SVP + kc], half);
#pragma unroll
      for (int j = 0; j < 4; ++j)
        acc[j] = __builtin_amdgcn_wmma_f32_16x16x32_bf16(
            false, a, false, bfr[j], (short)0, acc[j], false, false);
    }

    if (t + 1 < S_)   // warm L2 for next step's x_proj row
      __builtin_prefetch(
          &xproj[((size_t)(t + 1) * B_ + mt * 16 + nn) * FOURD + dcol], 0, 0);

#pragma unroll
    for (int r = 0; r < 8; ++r) {
      const int bb = mt * 16 + r + 8 * half;
      const size_t xr = ((size_t)t * B_ + bb) * FOURD;
      const float gi = sigmoidf_(acc[0][r] + xproj[xr + 0 * D_ + dcol]);
      const float gf = sigmoidf_(acc[1][r] + xproj[xr + 1 * D_ + dcol]);
      const float gg = tanhf    (acc[2][r] + xproj[xr + 2 * D_ + dcol]);
      const float go = sigmoidf_(acc[3][r] + xproj[xr + 3 * D_ + dcol]);
      const int ci = bb * D_ + dcol;
      const float cn = gf * cbuf[ci] + gi * gg;
      const float hn = go * tanhf(cn);
      cbuf[ci] = cn;
      out[((size_t)bb * S_ + t) * D_ + dcol] = hn;    // hidden_seq [B][S][D]
      hwrite[ci] = (__bf16)hn;
      if (t == S_ - 1) { out[HT_OFF + ci] = hn; out[CT_OFF + ci] = cn; }
    }

    // ---- device-wide step barrier ----
    __syncthreads();
    if (tid < 32) __builtin_amdgcn_s_cluster_barrier(); // NOP unless clustered
    if (tid == 0) {
      __threadfence();  // release h/c writes at device scope
      __hip_atomic_fetch_add(cnt, 1u, __ATOMIC_ACQ_REL, __HIP_MEMORY_SCOPE_AGENT);
      const unsigned target = 16u * (unsigned)(t + 1);  // monotonic, no reset race
      while (__hip_atomic_load(cnt, __ATOMIC_ACQUIRE, __HIP_MEMORY_SCOPE_AGENT) < target)
        __builtin_amdgcn_s_sleep(1);
    }
    __syncthreads();
  }
}

// ---------------------------------------------------------------------------
extern "C" void kernel_launch(void* const* d_in, const int* in_sizes, int n_in,
                              void* d_out, int out_size, void* d_ws, size_t ws_size,
                              hipStream_t stream) {
  const float* cas  = (const float*)d_in[0];   // [B][S][D]
  const float* soc  = (const float*)d_in[1];   // [B][S][D]
  const float* W    = (const float*)d_in[2];   // [D][4D]
  const float* U    = (const float*)d_in[3];   // [D][4D]
  const float* V    = (const float*)d_in[4];   // [D][4D]
  const float* bias = (const float*)d_in[5];   // [4D]
  float* out = (float*)d_out;

  char* ws = (char*)d_ws;
  float*    xproj = (float*)ws;
  __bf16*   Wt    = (__bf16*)(ws + XPROJ_BYTES);
  __bf16*   Ut    = (__bf16*)(ws + XPROJ_BYTES + WT_BYTES);
  __bf16*   hbf   = (__bf16*)(ws + XPROJ_BYTES + 2 * WT_BYTES);
  float*    cbuf  = (float*) (ws + XPROJ_BYTES + 2 * WT_BYTES + HBF_BYTES);
  unsigned* cnt   = (unsigned*)(ws + XPROJ_BYTES + 2 * WT_BYTES + HBF_BYTES + C_BYTES);

  lstm_prep<<<2048, 256, 0, stream>>>(W, U, Wt, Ut, hbf, cbuf, cnt);
  xproj_gemm<<<dim3(16, 512), 256, 0, stream>>>(cas, soc, Wt, Ut, bias, xproj);
  lstm_persistent<<<16, 256, (128 + 64) * 520 * 2, stream>>>(xproj, V, hbf, cbuf, out, cnt);

  (void)in_sizes; (void)n_in; (void)out_size; (void)ws_size;
}
